// GNNActor_55052890800722
// MI455X (gfx1250) — matrix operand order repaired and checked by reading, not compiled, not run on hardware
//
#include <hip/hip_runtime.h>

// GNNActor for MI455X (gfx1250): bf16 WMMA GEMMs (fp32 accumulate), bf16 edge
// gather + fp32 hardware atomics for scatter, wave32 shuffle reduction head.

typedef __bf16 bf16_t;
typedef __attribute__((ext_vector_type(16))) __bf16 v16bf;
typedef __attribute__((ext_vector_type(4)))  __bf16 v4bf;
typedef __attribute__((ext_vector_type(8)))  float  v8f;

static constexpr int NN = 49152;    // nodes
static constexpr int NE = 614400;   // edges
static constexpr int C  = 128;      // in channels
static constexpr int H  = 256;      // hidden

union AFrag { v16bf v; uint4 q[2]; };

// ---------------- utility kernels ----------------
__global__ void k_zero_f32(float* __restrict__ p, size_t n) {
  size_t i = (size_t)blockIdx.x * blockDim.x + threadIdx.x;
  if (i < n) p[i] = 0.0f;
}

__global__ void k_cvt_bf16(const float* __restrict__ x, bf16_t* __restrict__ y, size_t n) {
  size_t i = (size_t)blockIdx.x * blockDim.x + threadIdx.x;
  if (i < n) y[i] = (bf16_t)x[i];
}

// Wt[n*K + k] = W[k*NCols + n]  (bf16) — weights become row-major in K so a
// WMMA B-fragment is two contiguous 16B loads per lane.
__global__ void k_transpose_bf16(const float* __restrict__ W, bf16_t* __restrict__ Wt,
                                 int K, int NCols) {
  int idx = blockIdx.x * blockDim.x + threadIdx.x;
  if (idx >= K * NCols) return;
  int k = idx / NCols, n = idx % NCols;
  Wt[(size_t)n * K + k] = (bf16_t)W[idx];
}

__global__ void k_deg_count(const int* __restrict__ ei, float* __restrict__ deg, int E) {
  int e = blockIdx.x * blockDim.x + threadIdx.x;
  if (e < E) unsafeAtomicAdd(&deg[ei[E + e]], 1.0f);
}

__global__ void k_dinv(float* __restrict__ deg, int n) {
  int i = blockIdx.x * blockDim.x + threadIdx.x;
  if (i < n) deg[i] = rsqrtf(deg[i] + 1.0f);   // +1 for the self-loop
}

// ---------------- WMMA GEMM: D[M,NC] = A[M,K] x Bt[NC,K]^T ----------------
// block = 128 threads (4 waves), block tile = 64x64, wave tile = 16x64.
// EPI: 0 = store raw, 1 = +bias then leaky_relu(0.01).
template<int K, int EPI>
__global__ __launch_bounds__(128) void k_wmma_gemm(
    const bf16_t* __restrict__ A, const bf16_t* __restrict__ Bt,
    const float* __restrict__ bias, bf16_t* __restrict__ D, int NC)
{
  const int lane = threadIdx.x & 31;
  const int wave = threadIdx.x >> 5;
  const int hf   = lane >> 4;     // lane half (0: lanes 0-15, 1: lanes 16-31)
  const int lm   = lane & 15;
  const int row0 = blockIdx.y * 64 + wave * 16;
  const int col0 = blockIdx.x * 64;

  const v8f vzero = {0.f, 0.f, 0.f, 0.f, 0.f, 0.f, 0.f, 0.f};
  v8f acc[4];
#pragma unroll
  for (int t = 0; t < 4; ++t) acc[t] = vzero;

  for (int kb = 0; kb < K; kb += 32) {
    // A fragment (16x32 bf16): elems 0..7 -> K = kb + hf*8 + i,
    //                          elems 8..15 -> K = kb + 16 + hf*8 + (i-8)
    AFrag a;
    const bf16_t* ap = A + (size_t)(row0 + lm) * K + kb + hf * 8;
    a.q[0] = *reinterpret_cast<const uint4*>(ap);
    a.q[1] = *reinterpret_cast<const uint4*>(ap + 16);
#pragma unroll
    for (int t = 0; t < 4; ++t) {
      // B fragment (32x16 bf16): col = lm, same K order as A fragment
      AFrag b;
      const bf16_t* bp = Bt + (size_t)(col0 + t * 16 + lm) * K + kb + hf * 8;
      b.q[0] = *reinterpret_cast<const uint4*>(bp);
      b.q[1] = *reinterpret_cast<const uint4*>(bp + 16);
      acc[t] = __builtin_amdgcn_wmma_f32_16x16x32_bf16(
          false, a.v, false, b.v, (short)0, acc[t], false, false);
    }
  }

#pragma unroll
  for (int t = 0; t < 4; ++t) {
    const int col = col0 + t * 16 + lm;
    const float bv = (EPI == 1) ? bias[col] : 0.0f;
#pragma unroll
    for (int r = 0; r < 8; ++r) {               // D elem r: row = row0 + r + 8*hf
      const int row = row0 + hf * 8 + r;
      float v = acc[t][r] + bv;
      if (EPI == 1) v = (v > 0.0f) ? v : 0.01f * v;
      D[(size_t)row * NC + col] = (bf16_t)v;
    }
  }
}

// ---------------- edge scatter: accum[dst] += h[src] * dinv[s]*dinv[d] -----
__global__ __launch_bounds__(256) void k_edge_scatter(
    const int* __restrict__ ei, const bf16_t* __restrict__ h,
    const float* __restrict__ dinv, float* __restrict__ accum, int E)
{
  int gid = blockIdx.x * blockDim.x + threadIdx.x;
  int e = gid >> 5;                 // one wave32 per edge, 4 channels/lane
  if (e >= E) return;
  int lane = gid & 31;
  int s = ei[e];
  int d = ei[E + e];
  float norm = dinv[s] * dinv[d];
  v4bf hv = *reinterpret_cast<const v4bf*>(h + (size_t)s * C + lane * 4);
  float* ap = accum + (size_t)d * C + lane * 4;
  unsafeAtomicAdd(ap + 0, (float)hv[0] * norm);
  unsafeAtomicAdd(ap + 1, (float)hv[1] * norm);
  unsafeAtomicAdd(ap + 2, (float)hv[2] * norm);
  unsafeAtomicAdd(ap + 3, (float)hv[3] * norm);
}

// ------------- self-loop + bias + relu + residual, emit bf16 ---------------
__global__ void k_residual(const float* __restrict__ accum, const bf16_t* __restrict__ h,
                           const float* __restrict__ dinv, const float* __restrict__ state,
                           const float* __restrict__ bc, bf16_t* __restrict__ xout,
                           size_t total)
{
  size_t idx = (size_t)blockIdx.x * blockDim.x + threadIdx.x;
  if (idx >= total) return;
  size_t i = idx >> 7;              // / C
  int    c = (int)(idx & (C - 1));
  float di = dinv[i];
  float v = accum[idx] + (float)h[idx] * di * di + bc[c];
  v = v > 0.0f ? v : 0.0f;          // relu
  v += state[idx];                  // residual
  xout[idx] = (bf16_t)v;
}

// ------------- head: GEMV(H->1) + softplus + atomic global sum -------------
__global__ __launch_bounds__(256) void k_head(const bf16_t* __restrict__ y2,
    const float* __restrict__ W3, const float* __restrict__ b3,
    float* __restrict__ conc, float* __restrict__ sum, int rows)
{
  int gid = blockIdx.x * blockDim.x + threadIdx.x;
  int row = gid >> 5;               // one wave32 per output row
  if (row >= rows) return;
  int lane = gid & 31;
  const bf16_t* yp = y2 + (size_t)row * H;
  float acc = 0.0f;
#pragma unroll
  for (int j = 0; j < H / 32; ++j) {
    int k = lane + j * 32;
    acc += (float)yp[k] * W3[k];
  }
#pragma unroll
  for (int off = 16; off > 0; off >>= 1)
    acc += __shfl_xor(acc, off, 32);
  if (lane == 0) {
    float z = acc + b3[0];
    float sp = (z > 0.0f) ? z + log1pf(expf(-z)) : log1pf(expf(z)); // stable softplus
    conc[row] = sp;
    unsafeAtomicAdd(sum, sp);
  }
}

__global__ void k_norm(const float* __restrict__ conc, const float* __restrict__ sum,
                       float* __restrict__ out, int n)
{
  int i = blockIdx.x * blockDim.x + threadIdx.x;
  if (i < n) out[i] = conc[i] / (sum[0] + 1e-20f);
}

// ---------------------------------------------------------------------------
extern "C" void kernel_launch(void* const* d_in, const int* in_sizes, int n_in,
                              void* d_out, int out_size, void* d_ws, size_t ws_size,
                              hipStream_t stream) {
  const float* state = (const float*)d_in[0];
  const int*   ei    = (const int*)d_in[1];
  const float* Wc    = (const float*)d_in[2];
  const float* bc    = (const float*)d_in[3];
  const float* W1    = (const float*)d_in[4];
  const float* b1    = (const float*)d_in[5];
  const float* W2    = (const float*)d_in[6];
  const float* b2    = (const float*)d_in[7];
  const float* W3    = (const float*)d_in[8];
  const float* b3    = (const float*)d_in[9];
  float* out = (float*)d_out;

  char* p = (char*)d_ws;
  auto take = [&](size_t bytes) -> char* {
    char* r = p;
    p += (bytes + 255) & ~(size_t)255;
    return r;
  };
  float*  deg     = (float*)take(sizeof(float) * NN);          // -> dinv in place
  float*  sum     = (float*)take(sizeof(float));
  bf16_t* state_b = (bf16_t*)take(sizeof(bf16_t) * (size_t)NN * C);
  bf16_t* Wct     = (bf16_t*)take(sizeof(bf16_t) * C * C);
  bf16_t* W1t     = (bf16_t*)take(sizeof(bf16_t) * C * H);
  bf16_t* W2t     = (bf16_t*)take(sizeof(bf16_t) * H * H);
  bf16_t* hbuf    = (bf16_t*)take(sizeof(bf16_t) * (size_t)NN * C);
  float*  accum   = (float*)take(sizeof(float)  * (size_t)NN * C);
  bf16_t* xbuf    = (bf16_t*)take(sizeof(bf16_t) * (size_t)NN * C);
  bf16_t* y1      = (bf16_t*)take(sizeof(bf16_t) * (size_t)NN * H);
  bf16_t* y2      = (bf16_t*)take(sizeof(bf16_t) * (size_t)NN * H);
  float*  conc    = (float*)take(sizeof(float) * NN);

  const int T = 256;
  const size_t NC_total = (size_t)NN * C;

  // zero scratch that is accumulated into (fresh every call)
  k_zero_f32<<<(NN + T - 1) / T, T, 0, stream>>>(deg, NN);
  k_zero_f32<<<1, 1, 0, stream>>>(sum, 1);
  k_zero_f32<<<(unsigned)((NC_total + T - 1) / T), T, 0, stream>>>(accum, NC_total);

  // conversions / weight transposes (bf16, [N,K] layout for B fragments)
  k_cvt_bf16<<<(unsigned)((NC_total + T - 1) / T), T, 0, stream>>>(state, state_b, NC_total);
  k_transpose_bf16<<<(C * C + T - 1) / T, T, 0, stream>>>(Wc, Wct, C, C);
  k_transpose_bf16<<<(C * H + T - 1) / T, T, 0, stream>>>(W1, W1t, C, H);
  k_transpose_bf16<<<(H * H + T - 1) / T, T, 0, stream>>>(W2, W2t, H, H);

  // degree -> dinv
  k_deg_count<<<(NE + T - 1) / T, T, 0, stream>>>(ei, deg, NE);
  k_dinv<<<(NN + T - 1) / T, T, 0, stream>>>(deg, NN);

  // h = state @ Wc   (WMMA bf16)
  k_wmma_gemm<C, 0><<<dim3(C / 64, NN / 64), 128, 0, stream>>>(state_b, Wct, nullptr, hbuf, C);

  // scatter-add normalized messages over edges
  k_edge_scatter<<<(int)(((size_t)NE * 32 + T - 1) / T), T, 0, stream>>>(ei, hbuf, deg, accum, NE);

  // self-loop term + bias + relu + residual -> bf16 activations
  k_residual<<<(unsigned)((NC_total + T - 1) / T), T, 0, stream>>>(accum, hbuf, deg, state, bc,
                                                                  xbuf, NC_total);

  // MLP layers (WMMA bf16, fused bias + leaky_relu)
  k_wmma_gemm<C, 1><<<dim3(H / 64, NN / 64), 128, 0, stream>>>(xbuf, W1t, b1, y1, H);
  k_wmma_gemm<H, 1><<<dim3(H / 64, NN / 64), 128, 0, stream>>>(y1, W2t, b2, y2, H);

  // head: 256->1 GEMV + softplus + global sum, then normalize
  k_head<<<(int)(((size_t)NN * 32 + T - 1) / T), T, 0, stream>>>(y2, W3, b3, conc, sum, NN);
  k_norm<<<(NN + T - 1) / T, T, 0, stream>>>(conc, sum, out, NN);

  (void)in_sizes; (void)n_in; (void)out_size; (void)ws_size;
}
